// SelfDecoderBlock_33285996544218
// MI455X (gfx1250) — matrix operand order repaired and checked by reading, not compiled, not run on hardware
//
#include <hip/hip_runtime.h>

// Problem constants (match reference).
constexpr int B_ = 2, S_ = 2048, E_ = 1024, H_ = 16, D_ = 64, F_ = H_ * D_;

typedef __attribute__((ext_vector_type(16))) __bf16        bf16x16;
typedef __attribute__((ext_vector_type(8)))  float         f32x8;
typedef __attribute__((ext_vector_type(4)))  unsigned int  u32x4;

union Frag {
    bf16x16 v;
    u32x4   q[2];
};

__device__ __forceinline__ unsigned short f2bf(float f) {
    unsigned int u = __float_as_uint(f);
    u += 0x7fffu + ((u >> 16) & 1u);   // round-to-nearest-even
    return (unsigned short)(u >> 16);
}

__device__ __forceinline__ f32x8 zero8() {
    f32x8 z;
    #pragma unroll
    for (int i = 0; i < 8; ++i) z[i] = 0.0f;
    return z;
}

// A-fragment (16x32 bf16, MxK): lane m=l%16, half h=l/16.
// elems 0..7 -> k = kb + h*8 + i ; elems 8..15 -> k = kb + 16 + h*8 + (i-8).
__device__ __forceinline__ Frag load_a(const unsigned short* base, int ldk,
                                       int row0, int kb, int lane) {
    const int m = lane & 15, h = lane >> 4;
    const unsigned short* p = base + (size_t)(row0 + m) * ldk + kb + h * 8;
    Frag f;
    f.q[0] = *reinterpret_cast<const u32x4*>(p);
    f.q[1] = *reinterpret_cast<const u32x4*>(p + 16);
    return f;
}

// B-fragment from B^T stored row-major (N x K): lane n=l%16, half h=l/16,
// elems 0..7 -> k = kb + h*8 + i ; elems 8..15 -> k = kb + 16 + h*8 + (i-8)  (two b128 loads).
__device__ __forceinline__ Frag load_bt(const unsigned short* base, int ldk,
                                        int n0, int kb, int lane) {
    const int n = lane & 15, h = lane >> 4;
    const unsigned short* p = base + (size_t)(n0 + n) * ldk + kb + h * 8;
    Frag f;
    f.q[0] = *reinterpret_cast<const u32x4*>(p);
    f.q[1] = *reinterpret_cast<const u32x4*>(p + 16);
    return f;
}

__device__ __forceinline__ f32x8 wmma_bf16(const Frag& a, const Frag& b, f32x8 c) {
    return __builtin_amdgcn_wmma_f32_16x16x32_bf16(false, a.v, false, b.v,
                                                   (short)0, c, false, false);
}

// ---------------- prep kernels ----------------
__global__ void k_cvt(const float* __restrict__ src, unsigned short* __restrict__ dst, int n) {
    int i = blockIdx.x * blockDim.x + threadIdx.x;
    if (i < n) dst[i] = f2bf(src[i]);
}

// W: [H,E,D] f32 -> Wt: [H,D,E] bf16
__global__ void k_transpose_w(const float* __restrict__ W, unsigned short* __restrict__ Wt) {
    int i = blockIdx.x * blockDim.x + threadIdx.x;
    if (i >= H_ * E_ * D_) return;
    int d = i % D_;
    int e = (i / D_) % E_;
    int h = i / (E_ * D_);
    Wt[((size_t)h * D_ + d) * E_ + e] = f2bf(W[i]);
}

// Wo: [F,E] f32 -> Wot: [E,F] bf16
__global__ void k_transpose_wo(const float* __restrict__ Wo, unsigned short* __restrict__ Wot) {
    int i = blockIdx.x * blockDim.x + threadIdx.x;
    if (i >= F_ * E_) return;
    int e = i % E_;
    int f = i / E_;
    Wot[(size_t)e * F_ + f] = f2bf(Wo[i]);
}

// ---------------- stage 1: QKV projection ----------------
// One wave per (16-row tile over B*S, head). Q,K row-major [B,H,S,D] bf16;
// V stored pre-transposed [B,H,D,S] bf16 (B^T layout for the PV GEMM).
__global__ __launch_bounds__(32) void k_qkv(const unsigned short* __restrict__ xb,
                                            const unsigned short* __restrict__ Wqt,
                                            const unsigned short* __restrict__ Wkt,
                                            const unsigned short* __restrict__ Wvt,
                                            unsigned short* __restrict__ Qo,
                                            unsigned short* __restrict__ Ko,
                                            unsigned short* __restrict__ Vt) {
    const int lane = threadIdx.x;
    const int h    = blockIdx.x % H_;
    const int rt   = blockIdx.x / H_;
    const int row0 = rt * 16;            // global row over B*S
    const int b    = row0 / S_;
    const int s0   = row0 % S_;

    f32x8 accq[4], acck[4], accv[4];
    #pragma unroll
    for (int j = 0; j < 4; ++j) { accq[j] = zero8(); acck[j] = zero8(); accv[j] = zero8(); }

    const unsigned short* wq = Wqt + (size_t)h * D_ * E_;
    const unsigned short* wk = Wkt + (size_t)h * D_ * E_;
    const unsigned short* wv = Wvt + (size_t)h * D_ * E_;

    for (int e0 = 0; e0 < E_; e0 += 32) {
        Frag a = load_a(xb, E_, row0, e0, lane);
        #pragma unroll
        for (int j = 0; j < 4; ++j) {
            Frag bq = load_bt(wq, E_, j * 16, e0, lane);
            accq[j] = wmma_bf16(a, bq, accq[j]);
            Frag bk = load_bt(wk, E_, j * 16, e0, lane);
            acck[j] = wmma_bf16(a, bk, acck[j]);
            Frag bv = load_bt(wv, E_, j * 16, e0, lane);
            accv[j] = wmma_bf16(a, bv, accv[j]);
        }
    }

    const int n16 = lane & 15, half = lane >> 4;
    const size_t bh = (size_t)b * H_ + h;
    unsigned short* qdst = Qo + (bh * S_ + s0) * D_;
    unsigned short* kdst = Ko + (bh * S_ + s0) * D_;
    unsigned short* vdst = Vt + bh * D_ * S_;
    #pragma unroll
    for (int j = 0; j < 4; ++j) {
        #pragma unroll
        for (int r = 0; r < 8; ++r) {
            int m = half * 8 + r;
            int n = j * 16 + n16;
            qdst[(size_t)m * D_ + n]        = f2bf(accq[j][r]);
            kdst[(size_t)m * D_ + n]        = f2bf(acck[j][r]);
            vdst[(size_t)n * S_ + (s0 + m)] = f2bf(accv[j][r]);
        }
    }
}

// ---------------- stage 2: flash attention ----------------
// One wave per 16-query tile; streams keys in tiles of 32 with online softmax.
__global__ __launch_bounds__(32) void k_attn(const unsigned short* __restrict__ Qi,
                                             const unsigned short* __restrict__ Ki,
                                             const unsigned short* __restrict__ Vt,
                                             unsigned short* __restrict__ Cc) {
    __shared__ unsigned short p_lds[16 * 32];
    const int lane = threadIdx.x;
    const int qt   = blockIdx.x % (S_ / 16);
    const int bh   = blockIdx.x / (S_ / 16);
    const int s0   = qt * 16;

    const unsigned short* qbase = Qi + (size_t)bh * S_ * D_;
    const unsigned short* kbase = Ki + (size_t)bh * S_ * D_;
    const unsigned short* vbase = Vt + (size_t)bh * D_ * S_;

    Frag a0 = load_a(qbase, D_, s0, 0,  lane);  // d = 0..31
    Frag a1 = load_a(qbase, D_, s0, 32, lane);  // d = 32..63

    f32x8 acc[4];
    float mrow[8], lrow[8];
    #pragma unroll
    for (int j = 0; j < 4; ++j) acc[j] = zero8();
    #pragma unroll
    for (int r = 0; r < 8; ++r) { mrow[r] = -3.0e38f; lrow[r] = 0.0f; }

    const float inv_d = 1.0f / (float)D_;
    const int n16 = lane & 15, half = lane >> 4;

    for (int t0 = 0; t0 < S_; t0 += 32) {
        if (t0 + 32 < S_)   // hint next K tile into cache (global_prefetch_b8)
            __builtin_prefetch(kbase + (size_t)(t0 + 32 + lane) * D_, 0, 1);

        // scores tile [16 x 32], two 16-col chunks, K-dim = D in 2 WMMAs each
        f32x8 sc[2];
        #pragma unroll
        for (int c = 0; c < 2; ++c) {
            f32x8 s = zero8();
            Frag b0 = load_bt(kbase, D_, t0 + c * 16, 0,  lane);
            s = wmma_bf16(a0, b0, s);
            Frag b1 = load_bt(kbase, D_, t0 + c * 16, 32, lane);
            s = wmma_bf16(a1, b1, s);
            sc[c] = s;
        }

        float ps[2][8];
        #pragma unroll
        for (int r = 0; r < 8; ++r) {
            float v0 = sc[0][r] * inv_d, v1 = sc[1][r] * inv_d;
            float tm = fmaxf(v0, v1);
            #pragma unroll
            for (int mk = 1; mk < 16; mk <<= 1) tm = fmaxf(tm, __shfl_xor(tm, mk, 32));
            float nm   = fmaxf(mrow[r], tm);
            float corr = __expf(mrow[r] - nm);
            float p0 = __expf(v0 - nm), p1 = __expf(v1 - nm);
            float rs = p0 + p1;
            #pragma unroll
            for (int mk = 1; mk < 16; mk <<= 1) rs += __shfl_xor(rs, mk, 32);
            lrow[r] = lrow[r] * corr + rs;
            mrow[r] = nm;
            ps[0][r] = p0; ps[1][r] = p1;
            #pragma unroll
            for (int j = 0; j < 4; ++j) acc[j][r] *= corr;
        }

        // D-layout -> A-layout transpose of P through LDS
        #pragma unroll
        for (int c = 0; c < 2; ++c)
            #pragma unroll
            for (int r = 0; r < 8; ++r)
                p_lds[(half * 8 + r) * 32 + c * 16 + n16] = f2bf(ps[c][r]);
        asm volatile("s_wait_dscnt 0" ::: "memory");

        Frag pa;
        {
            const unsigned short* p = p_lds + (lane & 15) * 32 + (lane >> 4) * 8;
            pa.q[0] = *reinterpret_cast<const u32x4*>(p);
            pa.q[1] = *reinterpret_cast<const u32x4*>(p + 16);
        }

        #pragma unroll
        for (int j = 0; j < 4; ++j) {
            Frag vb = load_bt(vbase, S_, j * 16, t0, lane);  // V^T rows = d cols
            acc[j] = wmma_bf16(pa, vb, acc[j]);
        }
    }

    // normalize + store into concat layout [B, S, H*D] bf16
    const int hh = bh % H_;
    const int b  = bh / H_;
    unsigned short* dst = Cc + ((size_t)b * S_ + s0) * F_ + hh * D_;
    #pragma unroll
    for (int r = 0; r < 8; ++r) {
        float inv_l = 1.0f / lrow[r];
        int m = half * 8 + r;
        #pragma unroll
        for (int j = 0; j < 4; ++j)
            dst[(size_t)m * F_ + j * 16 + n16] = f2bf(acc[j][r] * inv_l);
    }
}

// ---------------- stage 3: output projection ----------------
// out[B*S, E] f32 = concat[B*S, F] @ Wo[F, E]  (Wo pre-transposed to [E,F])
__global__ __launch_bounds__(32) void k_proj(const unsigned short* __restrict__ Cc,
                                             const unsigned short* __restrict__ Wot,
                                             float* __restrict__ out) {
    const int lane = threadIdx.x;
    const int nt   = blockIdx.x % (E_ / 64);
    const int rt   = blockIdx.x / (E_ / 64);
    const int row0 = rt * 16;
    const int col0 = nt * 64;

    f32x8 acc[4];
    #pragma unroll
    for (int j = 0; j < 4; ++j) acc[j] = zero8();

    for (int f0 = 0; f0 < F_; f0 += 32) {
        Frag a = load_a(Cc, F_, row0, f0, lane);
        #pragma unroll
        for (int j = 0; j < 4; ++j) {
            Frag bw = load_bt(Wot, F_, col0 + j * 16, f0, lane);
            acc[j] = wmma_bf16(a, bw, acc[j]);
        }
    }

    const int n16 = lane & 15, half = lane >> 4;
    #pragma unroll
    for (int r = 0; r < 8; ++r) {
        int m = half * 8 + r;
        #pragma unroll
        for (int j = 0; j < 4; ++j)
            out[(size_t)(row0 + m) * E_ + col0 + j * 16 + n16] = acc[j][r];
    }
}

extern "C" void kernel_launch(void* const* d_in, const int* in_sizes, int n_in,
                              void* d_out, int out_size, void* d_ws, size_t ws_size,
                              hipStream_t stream) {
    const float* x  = (const float*)d_in[0];
    const float* Wq = (const float*)d_in[1];
    const float* Wk = (const float*)d_in[2];
    const float* Wv = (const float*)d_in[3];
    const float* Wo = (const float*)d_in[4];
    float* out = (float*)d_out;

    // Workspace carve-up (all bf16 buffers, 16B-aligned sizes).
    char* ws = (char*)d_ws;
    const size_t sz_xb = (size_t)B_ * S_ * E_ * 2;       // 8 MB
    const size_t sz_w  = (size_t)H_ * D_ * E_ * 2;       // 2 MB each
    const size_t sz_wo = (size_t)E_ * F_ * 2;            // 2 MB
    const size_t sz_q  = (size_t)B_ * H_ * S_ * D_ * 2;  // 8 MB each

    unsigned short* xb  = (unsigned short*)(ws);
    unsigned short* Wqt = (unsigned short*)(ws + sz_xb);
    unsigned short* Wkt = (unsigned short*)(ws + sz_xb + sz_w);
    unsigned short* Wvt = (unsigned short*)(ws + sz_xb + 2 * sz_w);
    unsigned short* Wot = (unsigned short*)(ws + sz_xb + 3 * sz_w);
    unsigned short* Qb  = (unsigned short*)(ws + sz_xb + 3 * sz_w + sz_wo);
    unsigned short* Kb  = (unsigned short*)(ws + sz_xb + 3 * sz_w + sz_wo + sz_q);
    unsigned short* Vt  = (unsigned short*)(ws + sz_xb + 3 * sz_w + sz_wo + 2 * sz_q);
    unsigned short* Cc  = (unsigned short*)(ws + sz_xb + 3 * sz_w + sz_wo + 3 * sz_q);

    // prep
    {
        int n = B_ * S_ * E_;
        k_cvt<<<(n + 255) / 256, 256, 0, stream>>>(x, xb, n);
    }
    {
        int n = H_ * E_ * D_;
        k_transpose_w<<<(n + 255) / 256, 256, 0, stream>>>(Wq, Wqt);
        k_transpose_w<<<(n + 255) / 256, 256, 0, stream>>>(Wk, Wkt);
        k_transpose_w<<<(n + 255) / 256, 256, 0, stream>>>(Wv, Wvt);
    }
    {
        int n = F_ * E_;
        k_transpose_wo<<<(n + 255) / 256, 256, 0, stream>>>(Wo, Wot);
    }

    // stage 1: QKV projection  (B*S/16 row tiles x H heads)
    k_qkv<<<(B_ * S_ / 16) * H_, 32, 0, stream>>>(xb, Wqt, Wkt, Wvt, Qb, Kb, Vt);

    // stage 2: attention  (B*H x S/16 query tiles)
    k_attn<<<B_ * H_ * (S_ / 16), 32, 0, stream>>>(Qb, Kb, Vt, Cc);

    // stage 3: output projection  (B*S/16 row tiles x E/64 col groups)
    k_proj<<<(B_ * S_ / 16) * (E_ / 64), 32, 0, stream>>>(Cc, Wot, out);
}
